// OutputPPBlock_57707180589102
// MI455X (gfx1250) — matrix operand order repaired and checked by reading, not compiled, not run on hardware
//
#include <hip/hip_runtime.h>
#include <hip/hip_fp16.h>

// ---------------------------------------------------------------------------
// Problem constants (match reference)
// ---------------------------------------------------------------------------
#define NNODES  50000
#define HDIM    128
#define DDIM    256
#define RDIM    6
#define LLAYERS 3
#define WAVES_PER_BLOCK 2          // 2 waves/block, each wave owns 32 nodes
#define NTILES16 (NNODES / 16)     // 3125

typedef __attribute__((ext_vector_type(16))) _Float16 v16h;
typedef __attribute__((ext_vector_type(8)))  _Float16 v8h;
typedef __attribute__((ext_vector_type(4)))  _Float16 v4h;
typedef __attribute__((ext_vector_type(8)))  float    v8f;
typedef __attribute__((ext_vector_type(4)))  float    f32x4;

// ---------------------------------------------------------------------------
// Utility kernels
// ---------------------------------------------------------------------------
__global__ void zero_f32(float* __restrict__ p, int n) {
    int i = blockIdx.x * blockDim.x + threadIdx.x;
    int stride = gridDim.x * blockDim.x;
    for (; i < n; i += stride) p[i] = 0.f;
}

__global__ void cvt_f32_to_f16(const float* __restrict__ src, _Float16* __restrict__ dst, int n) {
    int i = blockIdx.x * blockDim.x + threadIdx.x;
    int stride = gridDim.x * blockDim.x;
    for (; i < n; i += stride) dst[i] = (_Float16)src[i];
}

// W_out [1][256] -> padded f16 [16][256], row 0 = weights, rows 1..15 = 0.
// Lets the final C=1 projection run on the WMMA pipe instead of scalar VALU.
__global__ void prep_wout(const float* __restrict__ src, _Float16* __restrict__ dst) {
    int i = blockIdx.x * blockDim.x + threadIdx.x;
    if (i < 16 * DDIM) dst[i] = (i < DDIM) ? (_Float16)src[i] : (_Float16)0.f;
}

// ---------------------------------------------------------------------------
// Edge phase: h = (rbf @ W_rbf^T) * x, scatter-add into agg[N,H].
// One wave per edge; lane handles 4 consecutive H channels.
// x is a 512MB one-shot stream -> non-temporal loads so L2 stays dedicated to
// the agg atomic target (25.6MB) and the weights.
// ---------------------------------------------------------------------------
__global__ void edge_gate_scatter(const float* __restrict__ x,
                                  const float* __restrict__ rbf,
                                  const int*   __restrict__ idx,
                                  const float* __restrict__ Wr,   // [H][R]
                                  float*       __restrict__ agg,  // [N][H]
                                  int E) {
    int t = blockIdx.x * blockDim.x + threadIdx.x;
    int e = t >> 5;
    if (e >= E) return;
    int lane = t & 31;

    const float r0 = rbf[e * RDIM + 0], r1 = rbf[e * RDIM + 1], r2 = rbf[e * RDIM + 2];
    const float r3 = rbf[e * RDIM + 3], r4 = rbf[e * RDIM + 4], r5 = rbf[e * RDIM + 5];

    const f32x4 xv = __builtin_nontemporal_load(
        (const f32x4*)(x + (size_t)e * HDIM + lane * 4));

    const int node = idx[e];
    float* dst = agg + (size_t)node * HDIM + lane * 4;

#pragma unroll
    for (int j = 0; j < 4; ++j) {
        const float* w = Wr + (size_t)(lane * 4 + j) * RDIM;  // tiny table, L0-cached
        float g = w[0]*r0 + w[1]*r1 + w[2]*r2 + w[3]*r3 + w[4]*r4 + w[5]*r5;
        atomicAdd(dst + j, g * xv[j]);
    }
}

// ---------------------------------------------------------------------------
// WMMA helpers (wave32; this code's contiguous k order, shared by A and B)
// A 16x32 f16 fragment: lane m=lane&15, half=lane>>4;
//   elems 0..7  <- K = k0 + half*16 + e       (16B contiguous)
//   elems 8..15 <- K = k0 + half*16 + 8 + e   (16B contiguous)
// ---------------------------------------------------------------------------
__device__ __forceinline__ v16h load_a_lds(const _Float16* st, int m, int half, int k0) {
    v8h lo = *(const v8h*)(st + m * DDIM + k0 + half * 16);
    v8h hi = *(const v8h*)(st + m * DDIM + k0 + half * 16 + 8);
    v16h a;
#pragma unroll
    for (int e = 0; e < 8; ++e) { a[e] = lo[e]; a[e + 8] = hi[e]; }
    return a;
}

// One dense layer over TWO 16-row M-tiles (32 nodes):
//   sout[32 x 256] = act( sin[32 x K] @ W^T + b )
// Each B fragment is loaded once and fed to two WMMAs -> halves L2 weight traffic.
template <int K, bool SILU, bool BIAS>
__device__ __forceinline__ void mlp_layer2(const _Float16* __restrict__ sin_,
                                           _Float16*       __restrict__ sout,
                                           const _Float16* __restrict__ W,    // [256][K] f16
                                           const float*    __restrict__ bias, // [256] or null
                                           int lane) {
    const int m    = lane & 15;
    const int half = lane >> 4;
    constexpr int KT = K / 32;

    v16h af0[KT], af1[KT];
#pragma unroll
    for (int kt = 0; kt < KT; ++kt) {
        af0[kt] = load_a_lds(sin_,             m, half, kt * 32);  // rows 0..15
        af1[kt] = load_a_lds(sin_ + 16 * DDIM, m, half, kt * 32);  // rows 16..31
    }

#pragma unroll
    for (int nt = 0; nt < DDIM / 16; ++nt) {
        const int n = nt * 16 + m;  // this lane's output column
        v8f acc0 = {}, acc1 = {};
#pragma unroll
        for (int kt = 0; kt < KT; ++kt) {
            v16h b = *(const v16h*)(W + (size_t)n * K + kt * 32 + half * 16);
            acc0 = __builtin_amdgcn_wmma_f32_16x16x32_f16(false, af0[kt], false, b,
                                                          (short)0, acc0, false, false);
            acc1 = __builtin_amdgcn_wmma_f32_16x16x32_f16(false, af1[kt], false, b,
                                                          (short)0, acc1, false, false);
        }
        float bv = 0.f;
        if constexpr (BIAS) bv = bias[n];
#pragma unroll
        for (int r = 0; r < 8; ++r) {
            float v0 = acc0[r], v1 = acc1[r];
            if constexpr (BIAS) { v0 += bv; v1 += bv; }
            if constexpr (SILU) {
                v0 = v0 * (1.f / (1.f + __expf(-v0)));
                v1 = v1 * (1.f / (1.f + __expf(-v1)));
            }
            const int mo = r + half * 8;                  // C row within tile
            sout[mo * DDIM + n]        = (_Float16)v0;
            sout[(16 + mo) * DDIM + n] = (_Float16)v1;
        }
    }
    asm volatile("s_wait_dscnt 0" ::: "memory");
}

// ---------------------------------------------------------------------------
// Fused node pipeline: up-proj (128->256) + 3x SiLU MLP (256->256) + WMMA
// output projection (256->1). One wave per pair of 16-node tiles; ping-pong
// f16 staging in dynamic LDS (16KB x2 per wave; WGP has 320KB).
// ---------------------------------------------------------------------------
__global__ void node_mlp(const float*    __restrict__ agg,    // [N][H] f32
                         const _Float16* __restrict__ Wup16,  // [D][H] f16
                         const _Float16* __restrict__ Ws16,   // [L][D][D] f16
                         const float*    __restrict__ bs,     // [L][D] f32
                         const _Float16* __restrict__ Wout16, // [16][D] f16 (row0=W_out)
                         float*          __restrict__ out,    // [N] f32
                         int npairs) {
    extern __shared__ _Float16 smem[];
    const int wib  = threadIdx.x >> 5;
    const int lane = threadIdx.x & 31;
    const int pair = blockIdx.x * WAVES_PER_BLOCK + wib;
    if (pair >= npairs) return;  // whole-wave uniform exit (EXEC all-1 for WMMA)

    const int tile0 = pair * 2;
    const int tile1 = (tile0 + 1 < NTILES16) ? tile0 + 1 : tile0;  // tail: duplicate work

    _Float16* sA = smem + (size_t)wib * 2 * 32 * DDIM;
    _Float16* sB = sA + 32 * DDIM;

    // Stage the 32x128 f32 agg rows into LDS as f16 (row stride DDIM).
    // Vectorized: each lane moves a 4-float chunk per step (b128 load, b64 store).
#pragma unroll
    for (int it = 0; it < 32; ++it) {
        const int c  = it * 32 + lane;        // chunk id (4 floats each), 0..1023
        const int rr = c >> 5;                // 32 chunks per 128-wide row
        const int kk = (c & 31) * 4;
        const int t  = (rr < 16) ? tile0 : tile1;
        const int node = t * 16 + (rr & 15);
        const f32x4 v = *(const f32x4*)(agg + (size_t)node * HDIM + kk);
        v4h h;
        h[0] = (_Float16)v[0]; h[1] = (_Float16)v[1];
        h[2] = (_Float16)v[2]; h[3] = (_Float16)v[3];
        *(v4h*)(sA + rr * DDIM + kk) = h;
    }
    asm volatile("s_wait_dscnt 0" ::: "memory");

    mlp_layer2<HDIM, false, false>(sA, sB, Wup16, nullptr, lane);              // up-proj
    mlp_layer2<DDIM, true,  true >(sB, sA, Ws16 + 0 * DDIM * DDIM, bs + 0 * DDIM, lane);
    mlp_layer2<DDIM, true,  true >(sA, sB, Ws16 + 1 * DDIM * DDIM, bs + 1 * DDIM, lane);
    mlp_layer2<DDIM, true,  true >(sB, sA, Ws16 + 2 * DDIM * DDIM, bs + 2 * DDIM, lane);

    // Final projection to C=1 on the WMMA pipe: B = padded [16][256] W_out.
    // Valid results land in column n=0 (lanes 0 and 16).
    {
        const int m = lane & 15, half = lane >> 4;
        v8f acc0 = {}, acc1 = {};
#pragma unroll
        for (int kt = 0; kt < DDIM / 32; ++kt) {
            v16h a0 = load_a_lds(sA,             m, half, kt * 32);
            v16h a1 = load_a_lds(sA + 16 * DDIM, m, half, kt * 32);
            v16h b  = *(const v16h*)(Wout16 + (size_t)m * DDIM + kt * 32 + half * 16);
            acc0 = __builtin_amdgcn_wmma_f32_16x16x32_f16(false, a0, false, b,
                                                          (short)0, acc0, false, false);
            acc1 = __builtin_amdgcn_wmma_f32_16x16x32_f16(false, a1, false, b,
                                                          (short)0, acc1, false, false);
        }
        if (m == 0) {  // lanes 0 (rows 0..7) and 16 (rows 8..15) hold column 0
#pragma unroll
            for (int r = 0; r < 8; ++r) {
                out[tile0 * 16 + half * 8 + r] = acc0[r];
                out[tile1 * 16 + half * 8 + r] = acc1[r];  // tail: identical rewrite
            }
        }
    }
}

// ---------------------------------------------------------------------------
// Launch: zero agg -> convert weights -> edge scatter -> fused node pipeline
// ---------------------------------------------------------------------------
extern "C" void kernel_launch(void* const* d_in, const int* in_sizes, int n_in,
                              void* d_out, int out_size, void* d_ws, size_t ws_size,
                              hipStream_t stream) {
    const float* x     = (const float*)d_in[0];
    const float* rbf   = (const float*)d_in[1];
    const int*   idx   = (const int*)  d_in[2];
    const float* W_rbf = (const float*)d_in[3];
    const float* W_up  = (const float*)d_in[4];
    const float* Ws    = (const float*)d_in[5];
    const float* bs    = (const float*)d_in[6];
    const float* W_out = (const float*)d_in[7];
    const int E = in_sizes[0] / HDIM;

    // Workspace layout
    char* ws = (char*)d_ws;
    float* agg = (float*)ws;                                   // N*H f32 = 25.6 MB
    size_t off = (size_t)NNODES * HDIM * sizeof(float);
    _Float16* Wup16 = (_Float16*)(ws + off);
    off += (size_t)DDIM * HDIM * sizeof(_Float16);             // 64 KB
    _Float16* Ws16 = (_Float16*)(ws + off);
    off += (size_t)LLAYERS * DDIM * DDIM * sizeof(_Float16);   // 384 KB
    _Float16* Wout16 = (_Float16*)(ws + off);                  // 8 KB padded

    zero_f32<<<2048, 256, 0, stream>>>(agg, NNODES * HDIM);
    cvt_f32_to_f16<<<128, 256, 0, stream>>>(W_up, Wup16, DDIM * HDIM);
    cvt_f32_to_f16<<<384, 256, 0, stream>>>(Ws, Ws16, LLAYERS * DDIM * DDIM);
    prep_wout<<<16, 256, 0, stream>>>(W_out, Wout16);

    {
        long long tot = (long long)E * 32;
        int blocks = (int)((tot + 255) / 256);
        edge_gate_scatter<<<blocks, 256, 0, stream>>>(x, rbf, idx, W_rbf, agg, E);
    }
    {
        const int npairs = (NTILES16 + 1) / 2;  // 1563 waves, 32 nodes each
        const int blocks = (npairs + WAVES_PER_BLOCK - 1) / WAVES_PER_BLOCK;
        const size_t shmem = (size_t)WAVES_PER_BLOCK * 2 * 32 * DDIM * sizeof(_Float16); // 64KB
        node_mlp<<<blocks, 32 * WAVES_PER_BLOCK, shmem, stream>>>(
            agg, Wup16, Ws16, bs, Wout16, (float*)d_out, npairs);
    }
}
